// Project_91293824844326
// MI455X (gfx1250) — matrix operand order repaired and checked by reading, not compiled, not run on hardware
//
#include <hip/hip_runtime.h>
#include <math.h>

// ---------------------------------------------------------------------------
// Problem constants (match reference)
// ---------------------------------------------------------------------------
#define BATCH 1024
#define DIM   2048
#define NEQ   512
#define NIN   1024
#define NC    1536    // NEQ + NIN
#define DL    3072    // DIM + NIN
#define NITER 20

typedef __bf16 bf16_t;
typedef __attribute__((ext_vector_type(16))) bf16_t v16bf;
typedef __attribute__((ext_vector_type(8)))  bf16_t v8bf;
typedef __attribute__((ext_vector_type(8)))  float  v8f;

__device__ inline v8f wmma_bf16(v16bf a, v16bf b, v8f c) {
    return __builtin_amdgcn_wmma_f32_16x16x32_bf16(false, a, false, b,
                                                   (short)0, c, false, false);
}

__device__ inline v16bf ld_frag(const bf16_t* p0, const bf16_t* p1) {
    v8bf a = *(const v8bf*)p0;
    v8bf b = *(const v8bf*)p1;
    return __builtin_shufflevector(a, b, 0,1,2,3,4,5,6,7,8,9,10,11,12,13,14,15);
}

__device__ inline void split_store(float v, bf16_t* hi, bf16_t* lo) {
    bf16_t h = (bf16_t)v;
    *hi = h;
    *lo = (bf16_t)(v - (float)h);
}

// CDNA5 async global->LDS copy (ASYNCcnt-tracked, bypasses VGPRs).
// lds_off = LDS byte address (low 32 bits of generic shared pointer).
__device__ __forceinline__ void async_b128(unsigned lds_off, const bf16_t* g) {
    asm volatile("global_load_async_to_lds_b128 %0, %1, off"
                 :: "v"(lds_off), "v"(g) : "memory");
}
__device__ __forceinline__ void wait_async_24() {
    asm volatile("s_wait_asynccnt 0x18" ::: "memory");
}
__device__ __forceinline__ void wait_async_16() {
    asm volatile("s_wait_asynccnt 0x10" ::: "memory");
}
__device__ __forceinline__ void wait_async_8() {
    asm volatile("s_wait_asynccnt 0x8" ::: "memory");
}
__device__ __forceinline__ void wait_async_0() {
    asm volatile("s_wait_asynccnt 0x0" ::: "memory");
}

// ---------------------------------------------------------------------------
// Generic NT GEMM: C(MxN) = A(MxK) * B(NxK)^T , operands split bf16 hi/lo,
// fp32 accumulation via 3 WMMA products (a_hi*b_hi + a_lo*b_hi + a_hi*b_lo).
// Macro tile 128x128, 256 threads (8 waves), wave tile 32x64 (2x4 WMMA).
// 4-stage circular LDS pipeline fed by GLOBAL_LOAD_ASYNC_TO_LDS_B128
// (8 issues/wave/stage; up to 3 stages = 24 instructions in flight).
// mode==1: add cvec[col] to C (row-broadcast bias).
// Requires M,N % 128 == 0, K % 32 == 0.
// ---------------------------------------------------------------------------
#define GTM 128
#define GTN 128
#define GTK 32
#define NSTG 4

__global__ __launch_bounds__(256)
void gemm_nt_bf16x3(const bf16_t* __restrict__ Ahi, const bf16_t* __restrict__ Alo,
                    const bf16_t* __restrict__ Bhi, const bf16_t* __restrict__ Blo,
                    float* __restrict__ C, int M, int N, int K,
                    const float* __restrict__ cvec, int mode)
{
    __shared__ bf16_t sAh[NSTG][GTM * GTK];
    __shared__ bf16_t sAl[NSTG][GTM * GTK];
    __shared__ bf16_t sBh[NSTG][GTN * GTK];
    __shared__ bf16_t sBl[NSTG][GTN * GTK];

    const int tid  = threadIdx.x;
    const int lane = tid & 31;
    const int wave = tid >> 5;
    const int wm   = wave >> 1;          // 0..3 -> 32-row stripe
    const int wn   = wave & 1;           // 0..1 -> 64-col stripe
    const int m0   = blockIdx.y * GTM;
    const int n0   = blockIdx.x * GTN;

    const int halfsel = lane >> 4;       // 0: lanes 0-15, 1: lanes 16-31
    const int fr      = lane & 15;       // fragment row index
    const int akb     = halfsel * 8;     // A frag K base (ISA 16-bit A layout)
    const int bkb     = halfsel * 8;     // B frag K base (same layout as A)

    // Per-thread staging assignment: chunks tid and tid+256 of 512 x (8 bf16)
    const int r0 = tid >> 2;             // row of chunk 0 (chunk 1 = r0+64)
    const int q0 = (tid & 3) * 8;        // K sub-offset (elements)

    const unsigned offc0 = (unsigned)(r0 * GTK + q0) * 2u;   // bytes
    const unsigned offc1 = offc0 + (unsigned)(64 * GTK) * 2u;
    const unsigned bstr  = (unsigned)(GTM * GTK) * 2u;       // buffer stride
    const unsigned bAh = (unsigned)(uintptr_t)&sAh[0][0];
    const unsigned bAl = (unsigned)(uintptr_t)&sAl[0][0];
    const unsigned bBh = (unsigned)(uintptr_t)&sBh[0][0];
    const unsigned bBl = (unsigned)(uintptr_t)&sBl[0][0];

    const size_t gA0 = (size_t)(m0 + r0) * K + q0;
    const size_t gA1 = (size_t)(m0 + r0 + 64) * K + q0;
    const size_t gB0 = (size_t)(n0 + r0) * K + q0;
    const size_t gB1 = (size_t)(n0 + r0 + 64) * K + q0;

    auto issue = [&](int ks) {
        const int k0 = ks * GTK;
        const unsigned bo = (unsigned)(ks & (NSTG - 1)) * bstr;
        async_b128(bAh + bo + offc0, Ahi + gA0 + k0);
        async_b128(bAh + bo + offc1, Ahi + gA1 + k0);
        async_b128(bAl + bo + offc0, Alo + gA0 + k0);
        async_b128(bAl + bo + offc1, Alo + gA1 + k0);
        async_b128(bBh + bo + offc0, Bhi + gB0 + k0);
        async_b128(bBh + bo + offc1, Bhi + gB1 + k0);
        async_b128(bBl + bo + offc0, Blo + gB0 + k0);
        async_b128(bBl + bo + offc1, Blo + gB1 + k0);
    };

    v8f acc[2][4];
    #pragma unroll
    for (int i = 0; i < 2; i++)
        #pragma unroll
        for (int j = 0; j < 4; j++)
            #pragma unroll
            for (int r = 0; r < 8; r++)
                acc[i][j][r] = 0.0f;

    const int nstages = K / GTK;         // >= 48 for all call sites
    issue(0); issue(1); issue(2);        // prologue: 3 stages in flight

    for (int ks = 0; ks < nstages; ks++) {
        const int buf = ks & (NSTG - 1);
        const int rem = nstages - 1 - ks;
        if (ks + 3 < nstages) issue(ks + 3);
        // Exact retire point for stage ks: at most rem later stages in flight.
        if (rem >= 3)      wait_async_24();
        else if (rem == 2) wait_async_16();
        else if (rem == 1) wait_async_8();
        else               wait_async_0();
        __syncthreads();

        const bf16_t* cAh = &sAh[buf][0];
        const bf16_t* cAl = &sAl[buf][0];
        const bf16_t* cBh = &sBh[buf][0];
        const bf16_t* cBl = &sBl[buf][0];

        v16bf afh[2], afl[2], bfh[4], bfl[4];
        #pragma unroll
        for (int mi = 0; mi < 2; mi++) {
            const int r = wm * 32 + mi * 16 + fr;
            afh[mi] = ld_frag(&cAh[r * GTK + akb], &cAh[r * GTK + 16 + akb]);
            afl[mi] = ld_frag(&cAl[r * GTK + akb], &cAl[r * GTK + 16 + akb]);
        }
        #pragma unroll
        for (int ni = 0; ni < 4; ni++) {
            const int r = wn * 64 + ni * 16 + fr;
            bfh[ni] = ld_frag(&cBh[r * GTK + bkb], &cBh[r * GTK + 16 + bkb]);
            bfl[ni] = ld_frag(&cBl[r * GTK + bkb], &cBl[r * GTK + 16 + bkb]);
        }

        #pragma unroll
        for (int mi = 0; mi < 2; mi++)
            #pragma unroll
            for (int ni = 0; ni < 4; ni++) {
                acc[mi][ni] = wmma_bf16(afh[mi], bfh[ni], acc[mi][ni]);
                acc[mi][ni] = wmma_bf16(afl[mi], bfh[ni], acc[mi][ni]);
                acc[mi][ni] = wmma_bf16(afh[mi], bfl[ni], acc[mi][ni]);
            }
        __syncthreads();                 // buffer reuse fence
    }

    // Epilogue: C/D layout -> elem r maps to row (r + 8*halfsel), col = lane&15
    #pragma unroll
    for (int mi = 0; mi < 2; mi++)
        #pragma unroll
        for (int ni = 0; ni < 4; ni++) {
            const int col = n0 + wn * 64 + ni * 16 + fr;
            float bias = (mode == 1) ? cvec[col] : 0.0f;
            #pragma unroll
            for (int r = 0; r < 8; r++) {
                const int row = m0 + wm * 32 + mi * 16 + r + halfsel * 8;
                C[(size_t)row * N + col] = acc[mi][ni][r] + bias;
            }
        }
}

// ---------------------------------------------------------------------------
// Small helper kernels
// ---------------------------------------------------------------------------
__global__ __launch_bounds__(256)
void pack_W(const float* __restrict__ A, const float* __restrict__ G,
            bf16_t* Wh, bf16_t* Wl, bf16_t* Wth, bf16_t* Wtl)
{
    size_t i = (size_t)blockIdx.x * 256 + threadIdx.x;
    if (i >= (size_t)NC * DIM) return;
    int r = (int)(i / DIM), c = (int)(i % DIM);
    float v = (r < NEQ) ? A[(size_t)r * DIM + c] : G[(size_t)(r - NEQ) * DIM + c];
    split_store(v, &Wh[i], &Wl[i]);
    split_store(v, &Wth[(size_t)c * NC + r], &Wtl[(size_t)c * NC + r]);
}

__global__ __launch_bounds__(256)
void split_f32(const float* __restrict__ src, bf16_t* hi, bf16_t* lo, size_t n)
{
    size_t i = (size_t)blockIdx.x * 256 + threadIdx.x;
    if (i < n) split_store(src[i], &hi[i], &lo[i]);
}

__global__ __launch_bounds__(256)
void fill_zero_pair(float* p, bf16_t* hi, bf16_t* lo, size_t n)
{
    size_t i = (size_t)blockIdx.x * 256 + threadIdx.x;
    if (i < n) { p[i] = 0.0f; hi[i] = (bf16_t)0.0f; lo[i] = (bf16_t)0.0f; }
}

__global__ __launch_bounds__(256)
void add_diag(float* Mm)
{
    int i = blockIdx.x * 256 + threadIdx.x;
    if (i < NC) Mm[(size_t)i * NC + i] += 1e-6f + (i >= NEQ ? 1.0f : 0.0f);
}

// ---------------------------------------------------------------------------
// Blocked Cholesky (NB = 64), in-place lower factor of Mm (NC x NC)
// ---------------------------------------------------------------------------
#define CNB 64

__global__ __launch_bounds__(256)
void chol_diag(float* Mm, int kb)
{
    __shared__ float s[CNB * (CNB + 1)];
    const int o = kb * CNB;
    for (int i = threadIdx.x; i < CNB * CNB; i += 256) {
        int r = i >> 6, c = i & 63;
        s[r * (CNB + 1) + c] = Mm[(size_t)(o + r) * NC + o + c];
    }
    __syncthreads();
    for (int j = 0; j < CNB; j++) {
        if (threadIdx.x == 0) s[j * (CNB + 1) + j] = sqrtf(s[j * (CNB + 1) + j]);
        __syncthreads();
        float piv = s[j * (CNB + 1) + j];
        for (int i = j + 1 + threadIdx.x; i < CNB; i += 256)
            s[i * (CNB + 1) + j] /= piv;
        __syncthreads();
        for (int i = j + 1 + threadIdx.x; i < CNB; i += 256) {
            float lij = s[i * (CNB + 1) + j];
            for (int c = j + 1; c <= i; c++)
                s[i * (CNB + 1) + c] -= lij * s[c * (CNB + 1) + j];
        }
        __syncthreads();
    }
    for (int i = threadIdx.x; i < CNB * CNB; i += 256) {
        int r = i >> 6, c = i & 63;
        Mm[(size_t)(o + r) * NC + o + c] = s[r * (CNB + 1) + c];
    }
}

__global__ __launch_bounds__(64)
void chol_trsm(float* Mm, int kb)
{
    __shared__ float L[CNB * (CNB + 1)];
    __shared__ float xs[CNB * CNB];          // xs[j*64 + t]
    const int o = kb * CNB;
    const int t = threadIdx.x;
    for (int i = t; i < CNB * CNB; i += 64) {
        int r = i >> 6, c = i & 63;
        L[r * (CNB + 1) + c] = Mm[(size_t)(o + r) * NC + o + c];
    }
    __syncthreads();
    const int gr = o + CNB + blockIdx.x * CNB + t;
    for (int j = 0; j < CNB; j++) xs[j * CNB + t] = Mm[(size_t)gr * NC + o + j];
    for (int j = 0; j < CNB; j++) {
        float a = xs[j * CNB + t];
        for (int k = 0; k < j; k++) a -= L[j * (CNB + 1) + k] * xs[k * CNB + t];
        xs[j * CNB + t] = a / L[j * (CNB + 1) + j];
    }
    for (int j = 0; j < CNB; j++) Mm[(size_t)gr * NC + o + j] = xs[j * CNB + t];
}

__global__ __launch_bounds__(1024)
void syrk_update(float* Mm, int kb)
{
    __shared__ float sA[32 * 33];
    __shared__ float sB[32 * 33];
    const int o  = kb * CNB + CNB;
    const int ty = threadIdx.y, tx = threadIdx.x;
    const int gi = o + blockIdx.y * 32 + ty;
    const int gl = o + blockIdx.x * 32 + tx;
    float acc = 0.0f;
    for (int k0 = 0; k0 < CNB; k0 += 32) {
        sA[ty * 33 + tx] = Mm[(size_t)gi * NC + kb * CNB + k0 + tx];
        sB[ty * 33 + tx] = Mm[(size_t)(o + blockIdx.x * 32 + ty) * NC + kb * CNB + k0 + tx];
        __syncthreads();
        #pragma unroll
        for (int k = 0; k < 32; k++) acc += sA[ty * 33 + k] * sB[tx * 33 + k];
        __syncthreads();
    }
    Mm[(size_t)gi * NC + gl] -= acc;
}

// ---------------------------------------------------------------------------
// Triangular inverse: solve L * X = I ; write X^T rows (LinvT) fp32 + bf16 pair
// ---------------------------------------------------------------------------
__global__ __launch_bounds__(256)
void linv_kernel(const float* __restrict__ Lm, float* L32, bf16_t* Lh, bf16_t* Ll)
{
    int c = blockIdx.x * 256 + threadIdx.x;
    if (c >= NC) return;
    for (int j = 0; j < c; j++) {
        size_t idx = (size_t)c * NC + j;
        L32[idx] = 0.0f; Lh[idx] = (bf16_t)0.0f; Ll[idx] = (bf16_t)0.0f;
    }
    for (int j = c; j < NC; j++) {
        float v;
        if (j == c) {
            v = 1.0f / Lm[(size_t)j * NC + j];
        } else {
            float acc = 0.0f;
            for (int k = c; k < j; k++)
                acc += Lm[(size_t)j * NC + k] * L32[(size_t)c * NC + k];
            v = -acc / Lm[(size_t)j * NC + j];
        }
        size_t idx = (size_t)c * NC + j;
        L32[idx] = v;
        split_store(v, &Lh[idx], &Ll[idx]);
    }
}

// ---------------------------------------------------------------------------
// Matvecs for the constant vector c
// ---------------------------------------------------------------------------
__global__ __launch_bounds__(256)
void matvec_minv_f(const float* __restrict__ Minv, const float* __restrict__ bv,
                   const float* __restrict__ hv, float* __restrict__ u)
{
    int row  = blockIdx.x * 8 + (threadIdx.x >> 5);
    int lane = threadIdx.x & 31;
    if (row >= NC) return;
    float acc = 0.0f;
    for (int k = lane; k < NC; k += 32) {
        float f = (k < NEQ) ? bv[k] : hv[k - NEQ];
        acc += Minv[(size_t)row * NC + k] * f;
    }
    for (int off = 16; off; off >>= 1) acc += __shfl_down(acc, off, 32);
    if (lane == 0) u[row] = acc;
}

__global__ __launch_bounds__(256)
void matvec_c(const bf16_t* __restrict__ Wth, const bf16_t* __restrict__ Wtl,
              const float* __restrict__ u, float* __restrict__ cvec)
{
    int row  = blockIdx.x * 8 + (threadIdx.x >> 5);
    int lane = threadIdx.x & 31;
    if (row >= DL) return;
    if (row < DIM) {
        float acc = 0.0f;
        for (int k = lane; k < NC; k += 32) {
            float w = (float)Wth[(size_t)row * NC + k] + (float)Wtl[(size_t)row * NC + k];
            acc += w * u[k];
        }
        for (int off = 16; off; off >>= 1) acc += __shfl_down(acc, off, 32);
        if (lane == 0) cvec[row] = acc;
    } else if (lane == 0) {
        cvec[row] = u[NEQ + (row - DIM)];
    }
}

// ---------------------------------------------------------------------------
// Assemble P = I - E^T Minv E  as bf16 hi/lo (3072x3072)
// blocks: TL = W^T Minv W ; TR/BL from Tt = W^T Minv ; BR = Minv[512:,512:]
// ---------------------------------------------------------------------------
__global__ __launch_bounds__(256)
void assemble_P(const float* __restrict__ TLm, const float* __restrict__ Tt,
                const float* __restrict__ Minv, bf16_t* Ph, bf16_t* Pl)
{
    size_t idx = (size_t)blockIdx.x * 256 + threadIdx.x;
    if (idx >= (size_t)DL * DL) return;
    int a = (int)(idx / DL), b = (int)(idx % DL);
    float v;
    if (a < DIM) {
        v = (b < DIM) ? TLm[(size_t)a * DIM + b]
                      : Tt[(size_t)a * NC + NEQ + (b - DIM)];
    } else {
        v = (b < DIM) ? Tt[(size_t)b * NC + NEQ + (a - DIM)]
                      : Minv[(size_t)(NEQ + a - DIM) * NC + NEQ + (b - DIM)];
    }
    v = ((a == b) ? 1.0f : 0.0f) - v;
    split_store(v, &Ph[idx], &Pl[idx]);
}

// ---------------------------------------------------------------------------
// DR pointwise update (fused bf16 hi/lo split) and final interpolation
// ---------------------------------------------------------------------------
__global__ __launch_bounds__(256)
void dr_update_fused(float* __restrict__ y, const float* __restrict__ z,
                     const float* __restrict__ x, const float* __restrict__ lb,
                     const float* __restrict__ ub,
                     bf16_t* __restrict__ yh, bf16_t* __restrict__ yl)
{
    size_t i = (size_t)blockIdx.x * 256 + threadIdx.x;
    if (i >= (size_t)BATCH * DL) return;
    int col = (int)(i % DL);
    size_t row = i / DL;
    float zz = z[i], yy = y[i];
    float r = 2.0f * zz - yy, p;
    if (col < DIM) {
        float v = 0.5f * (r + x[row * DIM + col]);
        p = fminf(fmaxf(v, lb[col]), ub[col]);
    } else {
        p = fmaxf(r, 0.0f);
    }
    float yn = yy + p - zz;
    y[i] = yn;
    split_store(yn, &yh[i], &yl[i]);
}

__global__ __launch_bounds__(256)
void final_out(const float* __restrict__ z, const float* __restrict__ x,
               const float* __restrict__ iv, float* __restrict__ out)
{
    size_t i = (size_t)blockIdx.x * 256 + threadIdx.x;
    if (i >= (size_t)BATCH * DIM) return;
    size_t row = i / DIM;
    int col = (int)(i % DIM);
    float t = iv[0];
    out[i] = t * x[i] + (1.0f - t) * z[row * DL + col];
}

// ---------------------------------------------------------------------------
// Host orchestration
// ---------------------------------------------------------------------------
extern "C" void kernel_launch(void* const* d_in, const int* in_sizes, int n_in,
                              void* d_out, int out_size, void* d_ws, size_t ws_size,
                              hipStream_t stream)
{
    const float* x  = (const float*)d_in[0];
    const float* A  = (const float*)d_in[1];
    const float* bv = (const float*)d_in[2];
    const float* G  = (const float*)d_in[3];
    const float* hv = (const float*)d_in[4];
    const float* lb = (const float*)d_in[5];
    const float* ub = (const float*)d_in[6];
    const float* iv = (const float*)d_in[7];
    float* out = (float*)d_out;

    char* w = (char*)d_ws;
    size_t off = 0;
    auto alloc = [&](size_t bytes) { size_t r = off; off += (bytes + 255) & ~(size_t)255; return r; };

    const size_t oWH   = alloc((size_t)NC * DIM * 2);   // W hi  (bf16)
    const size_t oWL   = alloc((size_t)NC * DIM * 2);   // W lo
    const size_t oWTH  = alloc((size_t)DIM * NC * 2);   // W^T hi
    const size_t oWTL  = alloc((size_t)DIM * NC * 2);   // W^T lo
    const size_t oM    = alloc((size_t)NC * NC * 4);    // M / L (fp32)
    const size_t oLT32 = alloc((size_t)NC * NC * 4);    // LinvT fp32
    const size_t oLTH  = alloc((size_t)NC * NC * 2);    // LinvT hi
    const size_t oLTL  = alloc((size_t)NC * NC * 2);    // LinvT lo
    const size_t oMIH  = alloc((size_t)NC * NC * 2);    // Minv hi
    const size_t oMIL  = alloc((size_t)NC * NC * 2);    // Minv lo
    const size_t oMI   = alloc((size_t)NC * NC * 4);    // Minv fp32
    const size_t oTT   = alloc((size_t)DIM * NC * 4);   // Tt = W^T Minv (fp32)
    const size_t oTTH  = alloc((size_t)DIM * NC * 2);   // Tt hi
    const size_t oTTL  = alloc((size_t)DIM * NC * 2);   // Tt lo
    const size_t oTL   = alloc((size_t)DIM * DIM * 4);  // TL = W^T Minv W
    const size_t oPH   = alloc((size_t)DL * DL * 2);    // P hi
    const size_t oPL   = alloc((size_t)DL * DL * 2);    // P lo
    const size_t oU    = alloc((size_t)NC * 4);         // u = Minv f
    const size_t oC    = alloc((size_t)DL * 4);         // c vector
    (void)ws_size;

    // Buffer reuse (lifetimes are disjoint):
    float*  Y   = (float*)(w + oWH);    // over W hi+lo (dead after M GEMM)
    float*  Z   = (float*)(w + oM);     // over M + part LinvT32 (dead after Minv GEMM)
    bf16_t* YH  = (bf16_t*)(w + oLTH);  // over LinvT pair (dead after Minv GEMM)
    bf16_t* YL  = (bf16_t*)(w + oMIH);  // over Minv pair (dead after Tt GEMM)

    bf16_t* WH  = (bf16_t*)(w + oWH);   bf16_t* WL  = (bf16_t*)(w + oWL);
    bf16_t* WTH = (bf16_t*)(w + oWTH);  bf16_t* WTL = (bf16_t*)(w + oWTL);
    float*  Mm  = (float*)(w + oM);
    float*  LT32= (float*)(w + oLT32);
    bf16_t* LTH = (bf16_t*)(w + oLTH);  bf16_t* LTL = (bf16_t*)(w + oLTL);
    float*  MI  = (float*)(w + oMI);
    bf16_t* MIH = (bf16_t*)(w + oMIH);  bf16_t* MIL = (bf16_t*)(w + oMIL);
    float*  TT  = (float*)(w + oTT);
    bf16_t* TTH = (bf16_t*)(w + oTTH);  bf16_t* TTL = (bf16_t*)(w + oTTL);
    float*  TLm = (float*)(w + oTL);
    bf16_t* PH  = (bf16_t*)(w + oPH);   bf16_t* PL  = (bf16_t*)(w + oPL);
    float*  U   = (float*)(w + oU);
    float*  CV  = (float*)(w + oC);

    // 1) pack W = [A;G] as bf16 hi/lo, plus its transpose
    pack_W<<<((size_t)NC * DIM + 255) / 256, 256, 0, stream>>>(A, G, WH, WL, WTH, WTL);

    // 2) M = W W^T (+ diag)
    gemm_nt_bf16x3<<<dim3(NC / 128, NC / 128), 256, 0, stream>>>(
        WH, WL, WH, WL, Mm, NC, NC, DIM, nullptr, 0);
    add_diag<<<(NC + 255) / 256, 256, 0, stream>>>(Mm);

    // 3) blocked Cholesky in-place (24 blocks of 64)
    for (int kb = 0; kb < NC / CNB; kb++) {
        chol_diag<<<1, 256, 0, stream>>>(Mm, kb);
        int nb = NC / CNB - 1 - kb;
        if (nb > 0) {
            chol_trsm<<<nb, 64, 0, stream>>>(Mm, kb);
            syrk_update<<<dim3(nb * 2, nb * 2), dim3(32, 32), 0, stream>>>(Mm, kb);
        }
    }

    // 4) LinvT (solve L X = I), then Minv = LinvT * LinvT^T
    linv_kernel<<<NC / 256, 256, 0, stream>>>(Mm, LT32, LTH, LTL);
    gemm_nt_bf16x3<<<dim3(NC / 128, NC / 128), 256, 0, stream>>>(
        LTH, LTL, LTH, LTL, MI, NC, NC, NC, nullptr, 0);
    split_f32<<<((size_t)NC * NC + 255) / 256, 256, 0, stream>>>(
        MI, MIH, MIL, (size_t)NC * NC);

    // 5) Tt = W^T Minv ; TL = W^T Minv W
    gemm_nt_bf16x3<<<dim3(NC / 128, DIM / 128), 256, 0, stream>>>(
        WTH, WTL, MIH, MIL, TT, DIM, NC, NC, nullptr, 0);
    split_f32<<<((size_t)DIM * NC + 255) / 256, 256, 0, stream>>>(
        TT, TTH, TTL, (size_t)DIM * NC);
    gemm_nt_bf16x3<<<dim3(DIM / 128, DIM / 128), 256, 0, stream>>>(
        WTH, WTL, TTH, TTL, TLm, DIM, DIM, NC, nullptr, 0);

    // 6) constant vector c
    matvec_minv_f<<<NC / 8, 256, 0, stream>>>(MI, bv, hv, U);
    matvec_c<<<DL / 8, 256, 0, stream>>>(WTH, WTL, U, CV);

    // 7) P = I - E^T Minv E (bf16 hi/lo)
    assemble_P<<<((size_t)DL * DL + 255) / 256, 256, 0, stream>>>(TLm, TT, MI, PH, PL);

    // 8) DR iterations: z = y P + c ; fused pointwise update + bf16 split
    fill_zero_pair<<<((size_t)BATCH * DL + 255) / 256, 256, 0, stream>>>(
        Y, YH, YL, (size_t)BATCH * DL);
    for (int it = 0; it < NITER; it++) {
        gemm_nt_bf16x3<<<dim3(DL / 128, BATCH / 128), 256, 0, stream>>>(
            YH, YL, PH, PL, Z, BATCH, DL, DL, CV, 1);
        dr_update_fused<<<((size_t)BATCH * DL + 255) / 256, 256, 0, stream>>>(
            Y, Z, x, lb, ub, YH, YL);
    }

    // 9) final projection + interpolation with x
    gemm_nt_bf16x3<<<dim3(DL / 128, BATCH / 128), 256, 0, stream>>>(
        YH, YL, PH, PL, Z, BATCH, DL, DL, CV, 1);
    final_out<<<((size_t)BATCH * DIM + 255) / 256, 256, 0, stream>>>(Z, x, iv, out);
}
